// GNN_2619930051567
// MI455X (gfx1250) — matrix operand order repaired and checked by reading, not compiled, not run on hardware
//
#include <hip/hip_runtime.h>

#define D 128
#define WSTR 144  // padded LDS stride (bf16 elems) for transposed weights; 288B rows -> 32B aligned frags

typedef __attribute__((ext_vector_type(16))) __bf16 v16bf;
typedef __attribute__((ext_vector_type(8)))  float  v8f;

__device__ __forceinline__ __bf16 tobf(float f) { return (__bf16)f; }
__device__ __forceinline__ float sigm(float x) { return 1.0f / (1.0f + __expf(-x)); }
__device__ __forceinline__ float silu_f(float x) { return x * sigm(x); }

// ---------------------------------------------------------------------------
// elementwise helpers
// ---------------------------------------------------------------------------
__global__ void k_zero(float* __restrict__ p, int n) {
  int i = blockIdx.x * blockDim.x + threadIdx.x;
  if (i < n) p[i] = 0.0f;
}

__global__ void k_input_h(const float* __restrict__ x, const float* __restrict__ w,
                          const float* __restrict__ b, float* __restrict__ h, int n) {
  int i = blockIdx.x * blockDim.x + threadIdx.x;
  if (i >= n * D) return;
  int d = i & (D - 1); int r = i >> 7;
  float v = x[r * 2 + 0] * w[d] + x[r * 2 + 1] * w[D + d] + b[d];
  h[i] = silu_f(v);
}

__global__ void k_input_e(const float* __restrict__ ea, const float* __restrict__ w,
                          const float* __restrict__ b, float* __restrict__ e, int n) {
  int i = blockIdx.x * blockDim.x + threadIdx.x;
  if (i >= n * D) return;
  int d = i & (D - 1); int r = i >> 7;
  e[i] = silu_f(ea[r] * w[d] + b[d]);
}

__global__ void k_count(const int* __restrict__ src, float* __restrict__ cnt, int nE) {
  int i = blockIdx.x * blockDim.x + threadIdx.x;
  if (i < nE) atomicAdd(&cnt[src[i]], 1.0f);
}

// msg = sigmoid(e) * Vh[dst]; scatter-add to agg[src]
__global__ void k_edge_msg(const float* __restrict__ e, const float* __restrict__ Vh,
                           const int* __restrict__ src, const int* __restrict__ dst,
                           float* __restrict__ agg, int nE) {
  int i = blockIdx.x * blockDim.x + threadIdx.x;
  if (i >= nE * D) return;
  int d = i & (D - 1); int r = i >> 7;
  float m = sigm(e[i]) * Vh[(size_t)dst[r] * D + d];
  atomicAdd(&agg[(size_t)src[r] * D + d], m);
}

// ht = Uh + agg / max(cnt,1); accumulate per-column BN stats
__global__ void k_node_ht(const float* __restrict__ Uh, const float* __restrict__ agg,
                          const float* __restrict__ cnt, float* __restrict__ ht,
                          float* __restrict__ hsum, float* __restrict__ hsumsq, int nN) {
  int tid = blockIdx.x * blockDim.x + threadIdx.x;
  int col = tid & (D - 1);
  int rstep = (gridDim.x * blockDim.x) >> 7;
  float s = 0.f, s2 = 0.f;
  for (int row = tid >> 7; row < nN; row += rstep) {
    float c = cnt[row]; c = c < 1.0f ? 1.0f : c;
    float v = Uh[(size_t)row * D + col] + agg[(size_t)row * D + col] / c;
    ht[(size_t)row * D + col] = v;
    s += v; s2 += v * v;
  }
  atomicAdd(&hsum[col], s);
  atomicAdd(&hsumsq[col], s2);
}

// S layout: [0]=hsum [128]=hsumsq [256]=esum [384]=esumsq
//           [512]=hmean [640]=hrstd [768]=emean [896]=erstd
__global__ void k_finalize(float* __restrict__ S, float nN, float nE) {
  int t = threadIdx.x;
  if (t < 128) {
    float m = S[t] / nN;
    float v = S[128 + t] / nN - m * m;
    S[512 + t] = m;
    S[640 + t] = rsqrtf(v + 1e-5f);
  } else {
    int d = t - 128;
    float m = S[256 + d] / nE;
    float v = S[384 + d] / nE - m * m;
    S[768 + d] = m;
    S[896 + d] = rsqrtf(v + 1e-5f);
  }
}

// out += silu((t - mean) * rstd * gamma + beta)
__global__ void k_update(float* __restrict__ out, const float* __restrict__ t,
                         const float* __restrict__ mean, const float* __restrict__ rstd,
                         const float* __restrict__ gamma, const float* __restrict__ beta,
                         int n) {
  int i = blockIdx.x * blockDim.x + threadIdx.x;
  if (i >= n * D) return;
  int d = i & (D - 1);
  float v = (t[i] - mean[d]) * rstd[d] * gamma[d] + beta[d];
  out[i] += silu_f(v);
}

// out[row] = sigmoid(dot(z[row,:], w3) + b3)   (one wave per row)
__global__ __launch_bounds__(256)
void k_dot(const float* __restrict__ z, const float* __restrict__ w3,
           const float* __restrict__ b3, float* __restrict__ out, int rows) {
  int wave = (blockIdx.x * blockDim.x + threadIdx.x) >> 5;
  int lane = threadIdx.x & 31;
  int nw = (gridDim.x * blockDim.x) >> 5;
  for (int row = wave; row < rows; row += nw) {
    const float* zr = z + (size_t)row * D;
    float s = zr[lane] * w3[lane] + zr[lane + 32] * w3[lane + 32] +
              zr[lane + 64] * w3[lane + 64] + zr[lane + 96] * w3[lane + 96];
    s += __shfl_xor(s, 16);
    s += __shfl_xor(s, 8);
    s += __shfl_xor(s, 4);
    s += __shfl_xor(s, 2);
    s += __shfl_xor(s, 1);
    if (lane == 0) out[row] = sigm(s + b3[0]);
  }
}

// ---------------------------------------------------------------------------
// Generic WMMA GEMM, compile-time specialized:
//   out = maybe_silu( A[rows,128] @ W[128,128] + bias
//                     [+ G0[idx0[r]] + G1[idx1[r]]]  (GATHER) )
//   STATS: accumulate per-column sum / sumsq (pre-activation).
// rows must be a multiple of 16. Block = 256 threads = 8 waves; wave per tile.
// ---------------------------------------------------------------------------
template <bool GATHER, bool STATS, bool ACT>
__global__ __launch_bounds__(256)
void k_gemm(const float* __restrict__ A, const float* __restrict__ W,
            const float* __restrict__ bias, float* __restrict__ out, int rows,
            const float* __restrict__ G0, const int* __restrict__ idx0,
            const float* __restrict__ G1, const int* __restrict__ idx1,
            float* __restrict__ col_sum, float* __restrict__ col_sumsq) {
  __shared__ __align__(32) __bf16 wt[D * WSTR];  // transposed: wt[n*WSTR + k] = W[k][n]

  for (int i = threadIdx.x; i < D * D; i += 256) {
    int k = i >> 7, n = i & (D - 1);
    wt[n * WSTR + k] = tobf(W[i]);
  }
  __syncthreads();

  const int lane = threadIdx.x & 31;
  const int wave = threadIdx.x >> 5;
  const int half = lane >> 4;   // 0 or 1
  const int ln   = lane & 15;

  float sacc[8], s2acc[8];
#pragma unroll
  for (int i = 0; i < 8; ++i) { sacc[i] = 0.f; s2acc[i] = 0.f; }

  const int ntiles = rows >> 4;
  for (int tile = blockIdx.x * 8 + wave; tile < ntiles; tile += gridDim.x * 8) {
    const int rbase = tile << 4;

    // A fragments, 16x32 bf16, contiguous K order (matches B):
    // lane (half,ln): row M=ln; elems e -> K = kc*32 + half*16 + e
    v16bf afrag[4];
    const float* arow = A + (size_t)(rbase + ln) * D;
#pragma unroll
    for (int kc = 0; kc < 4; ++kc) {
      int k0 = kc * 32 + half * 16;
      const float4* p0 = (const float4*)(arow + k0);
      const float4* p1 = (const float4*)(arow + k0 + 8);
      float4 r0a = p0[0], r0b = p0[1];
      float4 r1a = p1[0], r1b = p1[1];
      v16bf a;
      a[0] = tobf(r0a.x);  a[1] = tobf(r0a.y);  a[2]  = tobf(r0a.z);  a[3]  = tobf(r0a.w);
      a[4] = tobf(r0b.x);  a[5] = tobf(r0b.y);  a[6]  = tobf(r0b.z);  a[7]  = tobf(r0b.w);
      a[8] = tobf(r1a.x);  a[9] = tobf(r1a.y);  a[10] = tobf(r1a.z);  a[11] = tobf(r1a.w);
      a[12] = tobf(r1b.x); a[13] = tobf(r1b.y); a[14] = tobf(r1b.z); a[15] = tobf(r1b.w);
      afrag[kc] = a;
    }

    // gather row bases (row-dependent), hoisted out of the nt loop
    size_t gb0[8], gb1[8];
    if (GATHER) {
#pragma unroll
      for (int r = 0; r < 8; ++r) {
        gb0[r] = (size_t)idx0[rbase + r + half * 8] * D;
        gb1[r] = (size_t)idx1[rbase + r + half * 8] * D;
      }
    }

#pragma unroll
    for (int nt = 0; nt < 8; ++nt) {
      const int col = nt * 16 + ln;
      const float bv = bias[col];
      v8f c;
#pragma unroll
      for (int i = 0; i < 8; ++i) c[i] = bv;

#pragma unroll
      for (int kc = 0; kc < 4; ++kc) {
        // B fragment 32x16: lane (half,ln): col N=ln, elems e -> K = kc*32 + half*16 + e
        const v16bf* bp = (const v16bf*)&wt[col * WSTR + kc * 32 + half * 16];
        c = __builtin_amdgcn_wmma_f32_16x16x32_bf16(false, afrag[kc], false, *bp,
                                                    (short)0, c, false, false);
      }

      // epilogue: C/D layout: VGPR r -> row r + 8*half, col = ln (within tile col nt)
      float g[8];
      if (GATHER) {
#pragma unroll
        for (int r = 0; r < 8; ++r) g[r] = G0[gb0[r] + col] + G1[gb1[r] + col];
      }
#pragma unroll
      for (int r = 0; r < 8; ++r) {
        int row = rbase + r + half * 8;
        float v = c[r];
        if (GATHER) v += g[r];
        if (STATS) { sacc[nt] += v; s2acc[nt] += v * v; }
        if (ACT) v = silu_f(v);
        out[(size_t)row * D + col] = v;
      }
    }
  }

  if (STATS) {
#pragma unroll
    for (int nt = 0; nt < 8; ++nt) {
      atomicAdd(&col_sum[nt * 16 + ln], sacc[nt]);
      atomicAdd(&col_sumsq[nt * 16 + ln], s2acc[nt]);
    }
  }
}

// ---------------------------------------------------------------------------
extern "C" void kernel_launch(void* const* d_in, const int* in_sizes, int n_in,
                              void* d_out, int out_size, void* d_ws, size_t ws_size,
                              hipStream_t stream) {
  (void)in_sizes; (void)n_in; (void)out_size; (void)ws_size;
  const int N = 20000, E = 320000, L = 3;

  const float* x    = (const float*)d_in[0];
  const float* ea   = (const float*)d_in[1];
  const int*   ei   = (const int*)d_in[2];
  const float* hpw  = (const float*)d_in[3];
  const float* hpb  = (const float*)d_in[4];
  const float* epw  = (const float*)d_in[5];
  const float* epb  = (const float*)d_in[6];
  const float* Uw   = (const float*)d_in[7];
  const float* Ub   = (const float*)d_in[8];
  const float* Vw   = (const float*)d_in[9];
  const float* Vb   = (const float*)d_in[10];
  const float* Aw   = (const float*)d_in[11];
  const float* Ab   = (const float*)d_in[12];
  const float* Bw   = (const float*)d_in[13];
  const float* Bb   = (const float*)d_in[14];
  const float* Cw   = (const float*)d_in[15];
  const float* Cb   = (const float*)d_in[16];
  const float* hg   = (const float*)d_in[17];
  const float* hbt  = (const float*)d_in[18];
  const float* eg   = (const float*)d_in[19];
  const float* ebt  = (const float*)d_in[20];
  const float* f1w  = (const float*)d_in[21];
  const float* f1b  = (const float*)d_in[22];
  const float* f2w  = (const float*)d_in[23];
  const float* f2b  = (const float*)d_in[24];
  const float* f3w  = (const float*)d_in[25];
  const float* f3b  = (const float*)d_in[26];
  const int* src = ei;       // scatter targets
  const int* dst = ei + E;   // gather sources

  float* ws = (float*)d_ws;
  size_t off = 0;
  float* h   = ws + off; off += (size_t)N * D;
  float* e   = ws + off; off += (size_t)E * D;
  float* Uh  = ws + off; off += (size_t)N * D;
  float* Vh  = ws + off; off += (size_t)N * D;
  float* Bh  = ws + off; off += (size_t)N * D;
  float* Ch  = ws + off; off += (size_t)N * D;
  float* agg = ws + off; off += (size_t)N * D;
  float* ht  = ws + off; off += (size_t)N * D;
  float* et  = ws + off; off += (size_t)E * D;
  float* cnt = ws + off; off += N;
  float* S   = ws + off; off += 1024;

  dim3 b256(256);
  int gN  = (N * D + 255) / 256;
  int gE  = (E * D + 255) / 256;
  int gNt = (N / 16 + 7) / 8;   // node GEMM tiles / 8 waves
  int gEt = (E / 16 + 7) / 8;   // edge GEMM tiles / 8 waves

  k_input_h<<<gN, b256, 0, stream>>>(x, hpw, hpb, h, N);
  k_input_e<<<gE, b256, 0, stream>>>(ea, epw, epb, e, E);
  k_zero<<<(N + 255) / 256, b256, 0, stream>>>(cnt, N);
  k_count<<<(E + 255) / 256, b256, 0, stream>>>(src, cnt, E);

  for (int l = 0; l < L; ++l) {
    size_t wo = (size_t)l * D * D, bo = (size_t)l * D;
    k_zero<<<(N * D + 255) / 256, b256, 0, stream>>>(agg, N * D);
    k_zero<<<2, b256, 0, stream>>>(S, 512);

    k_gemm<false, false, false><<<gNt, b256, 0, stream>>>(
        h, Uw + wo, Ub + bo, Uh, N,
        nullptr, nullptr, nullptr, nullptr, nullptr, nullptr);
    k_gemm<false, false, false><<<gNt, b256, 0, stream>>>(
        h, Vw + wo, Vb + bo, Vh, N,
        nullptr, nullptr, nullptr, nullptr, nullptr, nullptr);
    k_gemm<false, false, false><<<gNt, b256, 0, stream>>>(
        h, Bw + wo, Bb + bo, Bh, N,
        nullptr, nullptr, nullptr, nullptr, nullptr, nullptr);
    k_gemm<false, false, false><<<gNt, b256, 0, stream>>>(
        h, Cw + wo, Cb + bo, Ch, N,
        nullptr, nullptr, nullptr, nullptr, nullptr, nullptr);

    k_edge_msg<<<gE, b256, 0, stream>>>(e, Vh, src, dst, agg, E);

    // et = e@Aw + Ab + Bh[src] + Ch[dst], with fused BN stats
    k_gemm<true, true, false><<<gEt, b256, 0, stream>>>(
        e, Aw + wo, Ab + bo, et, E,
        Bh, src, Ch, dst, S + 256, S + 384);

    k_node_ht<<<128, b256, 0, stream>>>(Uh, agg, cnt, ht, S + 0, S + 128, N);
    k_finalize<<<1, b256, 0, stream>>>(S, (float)N, (float)E);
    k_update<<<gN, b256, 0, stream>>>(h, ht, S + 512, S + 640, hg + bo, hbt + bo, N);
    k_update<<<gE, b256, 0, stream>>>(e, et, S + 768, S + 896, eg + bo, ebt + bo, E);
  }

  // final MLP: z1 reuses et, z2 reuses e
  float* z1 = et;
  float* z2 = e;
  k_gemm<false, false, true><<<gEt, b256, 0, stream>>>(
      e, f1w, f1b, z1, E,
      nullptr, nullptr, nullptr, nullptr, nullptr, nullptr);
  k_gemm<false, false, true><<<gEt, b256, 0, stream>>>(
      z1, f2w, f2b, z2, E,
      nullptr, nullptr, nullptr, nullptr, nullptr, nullptr);
  k_dot<<<(E + 7) / 8, b256, 0, stream>>>(z2, f3w, f3b, (float*)d_out, E);
}
